// GraphAutoencoder_16578573762605
// MI455X (gfx1250) — matrix operand order repaired and checked by reading, not compiled, not run on hardware
//
#include <hip/hip_runtime.h>
#include <hip/hip_bf16.h>

// ---------------------------------------------------------------------------
// Problem constants (from reference)
// ---------------------------------------------------------------------------
#define TT   50
#define NA_  5632
#define ND_  5632
#define NB_  512
#define HH   256
#define NTOT (NA_ + ND_ + NB_)   // 11776
#define G3   768                 // 3*TH

typedef __attribute__((ext_vector_type(16))) __bf16 bf16x16;
typedef __attribute__((ext_vector_type(8)))  __bf16 bf16x8;
typedef __attribute__((ext_vector_type(8)))  float  f32x8;

// pack two fp32 -> packed bf16 pair (lowers to v_cvt_pk_bf16_f32)
__device__ __forceinline__ unsigned pack2_bf16(float a, float b)
{
    unsigned short ua = __builtin_bit_cast(unsigned short, (__bf16)a);
    unsigned short ub = __builtin_bit_cast(unsigned short, (__bf16)b);
    return (unsigned)ua | ((unsigned)ub << 16);
}

// contiguous 32B LDS fragment load as 2 x ds_load_b128, merged register-side
__device__ __forceinline__ bf16x16 lds_frag(const __bf16* p)
{
    bf16x8 lo = *(const bf16x8*)p;
    bf16x8 hi = *(const bf16x8*)(p + 8);
    return __builtin_shufflevector(lo, hi, 0, 1, 2, 3, 4, 5, 6, 7,
                                           8, 9, 10, 11, 12, 13, 14, 15);
}

// ---------------------------------------------------------------------------
// WMMA GEMM: C[M,N] (+)= A[M,K] * B  (+ bias)
//   A: fp32 row-major [M,K]  (converted to bf16 while staging into LDS)
//   B: fp32, BTRANS=false -> [K,N] row-major, BTRANS=true -> [N,K] row-major
//   C: fp32; row r maps to  C + (r/rowsPerBatch)*batchStrideC
//                              + (r%rowsPerBatch)*N + col
//   REQUIREMENT (all call sites satisfy it): rowsPerBatch % 128 == 0, so one
//   128-row block tile never crosses a batch boundary -> div/mod hoisted to
//   one scalar computation per block.
// Block: 128 threads = 4 wave32s; tile 128x64, BK=32.
// Each wave: 32 rows x 64 cols = 2x4 v_wmma_f32_16x16x32_bf16 per K-step.
// LDS tiles stored in WMMA *fragment order* so every lane's 16-element
// fragment is one contiguous 32-byte read (2 x ds_load_b128):
//   A: K position permuted  pos = (k&7) | ((k&8)<<1) | ((k&16)>>1)
//   B: stored N-major (Bs[n][k]); K permuted with the same pos as A
// Row stride 40 elems (80B): 16B-aligned fragments, conflict-free phases.
// ---------------------------------------------------------------------------
template<bool BTRANS, bool ACCUM>
__global__ __launch_bounds__(128)
void wmma_gemm(const float* __restrict__ A, const float* __restrict__ B,
               const float* __restrict__ bias, float* __restrict__ C,
               int M, int N, int K, int rowsPerBatch, long long batchStrideC)
{
    __shared__ alignas(16) __bf16 As[128][40];   // 128x32 tile (fragment order)
    __shared__ alignas(16) __bf16 Bs[64][40];    // 64(N) x 32(K) tile, N-major

    const int tid  = threadIdx.x;
    const int wave = tid >> 5;
    const int lane = tid & 31;
    const int r    = lane & 15;     // row/col within 16
    const int hb   = lane >> 4;     // lane half (ISA layout split)
    const int m0   = blockIdx.x * 128;
    const int n0   = blockIdx.y * 64;
    const bool fullM = (m0 + 128 <= M);   // uniform: no per-lane predication

    f32x8 acc[2][4];
#pragma unroll
    for (int mt = 0; mt < 2; ++mt)
#pragma unroll
        for (int nt = 0; nt < 4; ++nt)
#pragma unroll
            for (int i = 0; i < 8; ++i) acc[mt][nt][i] = 0.0f;

    for (int k0 = 0; k0 < K; k0 += 32) {
        // ---- stage A: 128 rows x 32 K = 2048 u32 writes / 128 thr ----
        if (fullM) {
#pragma unroll
            for (int i = 0; i < 16; ++i) {
                int idx = tid + (i << 7);
                int row = idx >> 4;
                int cp  = (idx & 15) << 1;
                const float* ap = A + (long long)(m0 + row) * K + k0 + cp;
                int pos = (cp & 7) | ((cp & 8) << 1) | ((cp & 16) >> 1);
                *(unsigned*)&As[row][pos] = pack2_bf16(ap[0], ap[1]);
            }
        } else {
#pragma unroll
            for (int i = 0; i < 16; ++i) {
                int idx = tid + (i << 7);
                int row = idx >> 4;
                int cp  = (idx & 15) << 1;
                int gr  = m0 + row;
                float a0 = 0.0f, a1 = 0.0f;
                if (gr < M) {
                    const float* ap = A + (long long)gr * K + k0 + cp;
                    a0 = ap[0]; a1 = ap[1];
                }
                int pos = (cp & 7) | ((cp & 8) << 1) | ((cp & 16) >> 1);
                *(unsigned*)&As[row][pos] = pack2_bf16(a0, a1);
            }
        }
        // ---- stage B: 64 n x 32 K = 1024 u32 writes / 128 thr ----
#pragma unroll
        for (int i = 0; i < 8; ++i) {
            int idx = tid + (i << 7);
            int n   = idx >> 4;
            int kp  = (idx & 15) << 1;
            float b0, b1;
            if (BTRANS) {
                const float* bp = B + (long long)(n0 + n) * K + k0 + kp;
                b0 = bp[0]; b1 = bp[1];
            } else {
                b0 = B[(long long)(k0 + kp) * N + n0 + n];
                b1 = B[(long long)(k0 + kp + 1) * N + n0 + n];
            }
            int kpos = (kp & 7) | ((kp & 8) << 1) | ((kp & 16) >> 1);
            *(unsigned*)&Bs[n][kpos] = pack2_bf16(b0, b1);
        }
        __syncthreads();

        bf16x16 af0 = lds_frag(&As[(wave << 5) + r][hb << 4]);
        bf16x16 af1 = lds_frag(&As[(wave << 5) + 16 + r][hb << 4]);
#pragma unroll
        for (int nt = 0; nt < 4; ++nt) {
            bf16x16 bf = lds_frag(&Bs[(nt << 4) + r][hb << 4]);
            acc[0][nt] = __builtin_amdgcn_wmma_f32_16x16x32_bf16(
                false, af0, false, bf, (short)0, acc[0][nt], false, false);
            acc[1][nt] = __builtin_amdgcn_wmma_f32_16x16x32_bf16(
                false, af1, false, bf, (short)0, acc[1][nt], false, false);
        }
        __syncthreads();
    }

    // ---- epilogue: C 16x16 f32 layout -> VGPR i: M = i + 8*hb, N = r ----
    // batch div/mod hoisted: whole 128-row tile lives in one batch slice
    const long long tb      = (long long)m0 / rowsPerBatch;
    const long long rowBase = (long long)m0 - tb * rowsPerBatch;
    float* Cblk = C + tb * batchStrideC + rowBase * (long long)N;
#pragma unroll
    for (int mt = 0; mt < 2; ++mt) {
#pragma unroll
        for (int nt = 0; nt < 4; ++nt) {
            int col = n0 + (nt << 4) + r;
            if (col >= N) continue;
            float bv = bias ? bias[col] : 0.0f;
#pragma unroll
            for (int i = 0; i < 8; ++i) {
                int lrow = (wave << 5) + (mt << 4) + (hb << 3) + i;
                if (m0 + lrow >= M) continue;
                float* cp = Cblk + (long long)lrow * N + col;
                float v = acc[mt][nt][i] + bv;
                if (ACCUM) *cp += v; else *cp = v;
            }
        }
    }
}

// ---------------------------------------------------------------------------
// Small elementwise / graph kernels
// ---------------------------------------------------------------------------
__global__ void proj_kernel(const float* __restrict__ x, const float* __restrict__ W,
                            const float* __restrict__ b, float* __restrict__ out,
                            long long rows, int F)
{
    long long idx = (long long)blockIdx.x * blockDim.x + threadIdx.x;
    if (idx >= rows * HH) return;
    long long i = idx >> 8; int c = (int)(idx & 255);
    float acc = b[c];
    for (int f = 0; f < F; ++f) acc += x[i * F + f] * W[f * HH + c];
    out[idx] = acc;
}

__global__ void degree_accum(const int* __restrict__ dst, float* __restrict__ deg, int E)
{
    int e = blockIdx.x * blockDim.x + threadIdx.x;
    if (e < E) atomicAdd(&deg[dst[e]], 1.0f);
}

__global__ void to_dinv(float* deg, int n)
{
    int i = blockIdx.x * blockDim.x + threadIdx.x;
    if (i < n) deg[i] = (deg[i] > 0.0f) ? rsqrtf(deg[i]) : 0.0f;
}

// z[t, nodeOff+n, c] = b1[c] (+ b2[c])  -- grid = T*Ntype blocks, 256 thr
__global__ void fill_bias2(float* __restrict__ z, const float* __restrict__ b1,
                           const float* __restrict__ b2, int Ntype, int nodeOff,
                           long long zStrideT)
{
    long long row = blockIdx.x;                 // t*Ntype + n
    int t = (int)(row / Ntype), n = (int)(row % Ntype);
    int c = threadIdx.x;
    float v = b1[c];
    if (b2) v += b2[c];
    z[(long long)t * zStrideT + (long long)(nodeOff + n) * HH + c] = v;
}

// GCN message scatter: z[t, off+dst, :] += h[t, src, :] * dinv[src]*dinv[dst]
__global__ void gcn_scatter(const float* __restrict__ h, const int* __restrict__ src,
                            const int* __restrict__ dst, const float* __restrict__ dinv,
                            float* __restrict__ z, int E, int Nsrc,
                            long long zStrideT, int nodeOff)
{
    int e = blockIdx.x % E;
    int t = blockIdx.x / E;
    int s = src[e], d = dst[e];
    float norm = dinv[s] * dinv[d];
    const float* hp = h + ((long long)t * Nsrc + s) * HH;
    float* zp = z + (long long)t * zStrideT + (long long)(nodeOff + d) * HH;
    atomicAdd(&zp[threadIdx.x], hp[threadIdx.x] * norm);
}

// SAGE aggregate: agg[t, dst, :] += xsrc[t, src, :]
__global__ void sage_scatter(const float* __restrict__ xsrc, const int* __restrict__ src,
                             const int* __restrict__ dst, float* __restrict__ agg,
                             int E, int Nsrc, int Ndst)
{
    int e = blockIdx.x % E;
    int t = blockIdx.x / E;
    int s = src[e], d = dst[e];
    const float* xp = xsrc + ((long long)t * Nsrc + s) * HH;
    float* ap = agg + ((long long)t * Ndst + d) * HH;
    atomicAdd(&ap[threadIdx.x], xp[threadIdx.x]);
}

// agg /= max(deg, 1) -- grid = T*Ndst blocks, 256 thr; agg contiguous [T*Ndst, H]
__global__ void mean_div(float* __restrict__ agg, const float* __restrict__ deg, int Ndst)
{
    long long row = blockIdx.x;
    int n = (int)(row % Ndst);
    float d = fmaxf(deg[n], 1.0f);
    agg[row * HH + threadIdx.x] /= d;
}

// GRU gate fusion over concatenated node state; grid = NTOT blocks, 256 thr
__global__ void gru_gate(const float* __restrict__ gi, const float* __restrict__ gh,
                         float* __restrict__ h)
{
    long long n = blockIdx.x;
    int c = threadIdx.x;
    const float* gin = gi + n * G3;
    const float* ghn = gh + n * G3;
    float r = 1.0f / (1.0f + __expf(-(gin[c]       + ghn[c])));
    float u = 1.0f / (1.0f + __expf(-(gin[HH + c]  + ghn[HH + c])));
    float g = tanhf(gin[2 * HH + c] + r * ghn[2 * HH + c]);
    long long idx = n * HH + c;
    h[idx] = (1.0f - u) * g + u * h[idx];
}

// Column means per node type -> hc[768]; grid = 768 blocks, 256 thr
__global__ void colmean(const float* __restrict__ h, float* __restrict__ hc)
{
    __shared__ float red[256];
    int b = blockIdx.x;
    int type = b >> 8, col = b & 255;
    int row0, cnt;
    if (type == 0)      { row0 = 0;         cnt = NA_; }
    else if (type == 1) { row0 = NA_;       cnt = ND_; }
    else                { row0 = NA_ + ND_; cnt = NB_; }
    float s = 0.0f;
    for (int r = threadIdx.x; r < cnt; r += 256)
        s += h[(long long)(row0 + r) * HH + col];
    red[threadIdx.x] = s;
    __syncthreads();
    for (int st = 128; st > 0; st >>= 1) {
        if (threadIdx.x < st) red[threadIdx.x] += red[threadIdx.x + st];
        __syncthreads();
    }
    if (threadIdx.x == 0) hc[type * HH + col] = red[0] / (float)cnt;
}

// out[o] = b_hp[o] + sum_j hc[j] * W_hp[j, o]   (1 block, 256 thr)
__global__ void final_proj(const float* __restrict__ hc, const float* __restrict__ Whp,
                           const float* __restrict__ bhp, float* __restrict__ out)
{
    int o = threadIdx.x;
    float acc = bhp[o];
    for (int j = 0; j < G3; ++j) acc += hc[j] * Whp[j * HH + o];
    out[o] = acc;
}

// ---------------------------------------------------------------------------
// Host orchestration
// ---------------------------------------------------------------------------
static inline void launch_gemm(const float* A, const float* B, const float* bias,
                               float* C, int M, int N, int K, int rowsPerBatch,
                               long long batchStrideC, bool btrans, bool accum,
                               hipStream_t stream)
{
    dim3 grid((M + 127) / 128, (N + 63) / 64), block(128);
    if (btrans) {
        if (accum) wmma_gemm<true,  true ><<<grid, block, 0, stream>>>(A, B, bias, C, M, N, K, rowsPerBatch, batchStrideC);
        else       wmma_gemm<true,  false><<<grid, block, 0, stream>>>(A, B, bias, C, M, N, K, rowsPerBatch, batchStrideC);
    } else {
        if (accum) wmma_gemm<false, true ><<<grid, block, 0, stream>>>(A, B, bias, C, M, N, K, rowsPerBatch, batchStrideC);
        else       wmma_gemm<false, false><<<grid, block, 0, stream>>>(A, B, bias, C, M, N, K, rowsPerBatch, batchStrideC);
    }
}

extern "C" void kernel_launch(void* const* d_in, const int* in_sizes, int n_in,
                              void* d_out, int out_size, void* d_ws, size_t ws_size,
                              hipStream_t stream)
{
    // ---- inputs (setup_inputs order) ----
    const float* x_attk = (const float*)d_in[0];
    const float* x_def  = (const float*)d_in[1];
    const float* x_ball = (const float*)d_in[2];
    const int* e_aa_src = (const int*)d_in[3];  const int* e_aa_dst = (const int*)d_in[4];
    const int* e_ad_src = (const int*)d_in[5];  const int* e_ad_dst = (const int*)d_in[6];
    const int* e_dd_src = (const int*)d_in[7];  const int* e_dd_dst = (const int*)d_in[8];
    const int* e_ab_src = (const int*)d_in[9];  const int* e_ab_dst = (const int*)d_in[10];
    const int* e_db_src = (const int*)d_in[11]; const int* e_db_dst = (const int*)d_in[12];
    const float* Wp_attk = (const float*)d_in[13]; const float* bp_attk = (const float*)d_in[14];
    const float* Wp_def  = (const float*)d_in[15]; const float* bp_def  = (const float*)d_in[16];
    const float* Wp_ball = (const float*)d_in[17]; const float* bp_ball = (const float*)d_in[18];
    const float* Wg_aa = (const float*)d_in[19]; const float* bg_aa = (const float*)d_in[20];
    const float* Wg_dd = (const float*)d_in[21]; const float* bg_dd = (const float*)d_in[22];
    const float* Wl_ad = (const float*)d_in[23]; const float* bl_ad = (const float*)d_in[24];
    const float* Wr_ad = (const float*)d_in[25];
    const float* Wl_ab = (const float*)d_in[26]; const float* bl_ab = (const float*)d_in[27];
    const float* Wr_ab = (const float*)d_in[28];
    const float* Wl_db = (const float*)d_in[29]; const float* bl_db = (const float*)d_in[30];
    const float* Wr_db = (const float*)d_in[31];
    const float* W_ih = (const float*)d_in[32]; const float* W_hh = (const float*)d_in[33];
    const float* b_ih = (const float*)d_in[34]; const float* b_hh = (const float*)d_in[35];
    const float* W_hp = (const float*)d_in[36]; const float* b_hp = (const float*)d_in[37];

    const int E_AA = in_sizes[3], E_AD = in_sizes[5], E_DD = in_sizes[7];
    const int E_AB = in_sizes[9], E_DB = in_sizes[11];

    const long long ZT = (long long)NTOT * HH;       // per-frame stride in z_all

    // ---- workspace bump allocator ----
    char* wsb = (char*)d_ws;
    size_t off = 0;
    auto alloc = [&](size_t nfloats) -> float* {
        float* p = (float*)(wsb + off);
        off = (off + nfloats * sizeof(float) + 255) & ~(size_t)255;
        return p;
    };
    float* pa     = alloc((size_t)TT * NA_ * HH);    // projected attackers
    float* pd     = alloc((size_t)TT * ND_ * HH);    // projected defenders
    float* pb     = alloc((size_t)TT * NB_ * HH);    // projected ball
    float* htmp   = alloc((size_t)TT * NA_ * HH);    // GCN lin output (reused)
    float* z_all  = alloc((size_t)TT * NTOT * HH);   // [T, 11776, 256]
    float* agg    = alloc((size_t)TT * ND_ * HH);    // SAGE mean buffer (reused)
    float* gi     = alloc((size_t)NTOT * G3);
    float* gh     = alloc((size_t)NTOT * G3);
    float* hstate = alloc((size_t)NTOT * HH);
    float* hc     = alloc(G3);
    float* dinv_a = alloc(NA_);
    float* dinv_d = alloc(ND_);
    float* deg_ad = alloc(ND_);
    float* deg_ab = alloc(NB_);
    float* deg_db = alloc(NB_);
    (void)ws_size; (void)n_in; (void)out_size;

    // ---- degrees ----
    hipMemsetAsync(dinv_a, 0, NA_ * sizeof(float), stream);
    hipMemsetAsync(dinv_d, 0, ND_ * sizeof(float), stream);
    hipMemsetAsync(deg_ad, 0, ND_ * sizeof(float), stream);
    hipMemsetAsync(deg_ab, 0, NB_ * sizeof(float), stream);
    hipMemsetAsync(deg_db, 0, NB_ * sizeof(float), stream);
    degree_accum<<<(E_AA + 255) / 256, 256, 0, stream>>>(e_aa_dst, dinv_a, E_AA);
    degree_accum<<<(E_DD + 255) / 256, 256, 0, stream>>>(e_dd_dst, dinv_d, E_DD);
    degree_accum<<<(E_AD + 255) / 256, 256, 0, stream>>>(e_ad_dst, deg_ad, E_AD);
    degree_accum<<<(E_AB + 255) / 256, 256, 0, stream>>>(e_ab_dst, deg_ab, E_AB);
    degree_accum<<<(E_DB + 255) / 256, 256, 0, stream>>>(e_db_dst, deg_db, E_DB);
    to_dinv<<<(NA_ + 255) / 256, 256, 0, stream>>>(dinv_a, NA_);
    to_dinv<<<(ND_ + 255) / 256, 256, 0, stream>>>(dinv_d, ND_);

    // ---- node projections ----
    proj_kernel<<<(unsigned)(((long long)TT * NA_ * HH + 255) / 256), 256, 0, stream>>>(
        x_attk, Wp_attk, bp_attk, pa, (long long)TT * NA_, 7);
    proj_kernel<<<(unsigned)(((long long)TT * ND_ * HH + 255) / 256), 256, 0, stream>>>(
        x_def, Wp_def, bp_def, pd, (long long)TT * ND_, 7);
    proj_kernel<<<(unsigned)(((long long)TT * NB_ * HH + 255) / 256), 256, 0, stream>>>(
        x_ball, Wp_ball, bp_ball, pb, (long long)TT * NB_, 4);

    // ---- seed z_all with the conv biases (everything else accumulates) ----
    fill_bias2<<<TT * NA_, 256, 0, stream>>>(z_all, bg_aa, nullptr, NA_, 0, ZT);
    fill_bias2<<<TT * ND_, 256, 0, stream>>>(z_all, bg_dd, bl_ad, ND_, NA_, ZT);
    fill_bias2<<<TT * NB_, 256, 0, stream>>>(z_all, bl_ab, bl_db, NB_, NA_ + ND_, ZT);

    // ---- GCN attk->attk: h = pa @ Wg_aa ; scatter-add normalized messages ----
    launch_gemm(pa, Wg_aa, nullptr, htmp, TT * NA_, HH, HH, TT * NA_, 0, false, false, stream);
    gcn_scatter<<<(unsigned)((long long)E_AA * TT), 256, 0, stream>>>(
        htmp, e_aa_src, e_aa_dst, dinv_a, z_all, E_AA, NA_, ZT, 0);

    // ---- GCN def->def ----
    launch_gemm(pd, Wg_dd, nullptr, htmp, TT * ND_, HH, HH, TT * ND_, 0, false, false, stream);
    gcn_scatter<<<(unsigned)((long long)E_DD * TT), 256, 0, stream>>>(
        htmp, e_dd_src, e_dd_dst, dinv_d, z_all, E_DD, ND_, ZT, NA_);

    // ---- SAGE attk->def: z_d += mean_agg(pa) @ Wl_ad + pd @ Wr_ad ----
    hipMemsetAsync(agg, 0, (size_t)TT * ND_ * HH * sizeof(float), stream);
    sage_scatter<<<(unsigned)((long long)E_AD * TT), 256, 0, stream>>>(
        pa, e_ad_src, e_ad_dst, agg, E_AD, NA_, ND_);
    mean_div<<<TT * ND_, 256, 0, stream>>>(agg, deg_ad, ND_);
    launch_gemm(agg, Wl_ad, nullptr, z_all + (long long)NA_ * HH,
                TT * ND_, HH, HH, ND_, ZT, false, true, stream);
    launch_gemm(pd, Wr_ad, nullptr, z_all + (long long)NA_ * HH,
                TT * ND_, HH, HH, ND_, ZT, false, true, stream);

    // ---- SAGE attk->ball: z_b += mean_agg(pa) @ Wl_ab + pb @ Wr_ab ----
    hipMemsetAsync(agg, 0, (size_t)TT * NB_ * HH * sizeof(float), stream);
    sage_scatter<<<(unsigned)((long long)E_AB * TT), 256, 0, stream>>>(
        pa, e_ab_src, e_ab_dst, agg, E_AB, NA_, NB_);
    mean_div<<<TT * NB_, 256, 0, stream>>>(agg, deg_ab, NB_);
    launch_gemm(agg, Wl_ab, nullptr, z_all + (long long)(NA_ + ND_) * HH,
                TT * NB_, HH, HH, NB_, ZT, false, true, stream);
    launch_gemm(pb, Wr_ab, nullptr, z_all + (long long)(NA_ + ND_) * HH,
                TT * NB_, HH, HH, NB_, ZT, false, true, stream);

    // ---- SAGE def->ball: z_b += mean_agg(pd) @ Wl_db + pb @ Wr_db ----
    hipMemsetAsync(agg, 0, (size_t)TT * NB_ * HH * sizeof(float), stream);
    sage_scatter<<<(unsigned)((long long)E_DB * TT), 256, 0, stream>>>(
        pd, e_db_src, e_db_dst, agg, E_DB, ND_, NB_);
    mean_div<<<TT * NB_, 256, 0, stream>>>(agg, deg_db, NB_);
    launch_gemm(agg, Wl_db, nullptr, z_all + (long long)(NA_ + ND_) * HH,
                TT * NB_, HH, HH, NB_, ZT, false, true, stream);
    launch_gemm(pb, Wr_db, nullptr, z_all + (long long)(NA_ + ND_) * HH,
                TT * NB_, HH, HH, NB_, ZT, false, true, stream);

    // ---- shared GRU over all 11776 nodes at once (weights shared) ----
    hipMemsetAsync(hstate, 0, (size_t)NTOT * HH * sizeof(float), stream);
    for (int t = 0; t < TT; ++t) {
        const float* zt = z_all + (long long)t * ZT;     // [NTOT, 256]
        launch_gemm(zt,     W_ih, b_ih, gi, NTOT, G3, HH, NTOT, 0, true, false, stream);
        launch_gemm(hstate, W_hh, b_hh, gh, NTOT, G3, HH, NTOT, 0, true, false, stream);
        gru_gate<<<NTOT, 256, 0, stream>>>(gi, gh, hstate);
    }

    // ---- mean-pool per type, final projection ----
    colmean<<<G3, 256, 0, stream>>>(hstate, hc);
    final_proj<<<1, 256, 0, stream>>>(hc, W_hp, b_hp, (float*)d_out);
}
